// ResidualBlock_26104811225176
// MI455X (gfx1250) — matrix operand order repaired and checked by reading, not compiled, not run on hardware
//
#include <hip/hip_runtime.h>
#include <hip/hip_bf16.h>

typedef _Float16 v16h __attribute__((ext_vector_type(16)));
typedef _Float16 v8h  __attribute__((ext_vector_type(8)));
typedef _Float16 v2h  __attribute__((ext_vector_type(2)));
typedef float    v8f  __attribute__((ext_vector_type(8)));

#define Bn 8
#define Cc 64
#define Hh 256
#define Ww 256
#define TR 8
#define TC 32
#define LROWS 10
#define LCOLS 34
#define PIXSTRIDE 72   // halves per pixel slot in LDS (144B, 16B aligned, conflict-free)

union V16 { v16h v; v8h h[2]; };

// CDNA5 async global->LDS copy path (guarded; falls back to plain copies).
#if defined(__has_builtin)
#if __has_builtin(__builtin_amdgcn_global_load_async_to_lds_b128) && \
    __has_builtin(__builtin_amdgcn_s_wait_asynccnt)
#define USE_ASYNC_LDS 1
#endif
#endif

#if defined(USE_ASYNC_LDS)
#define AS1 __attribute__((address_space(1)))
#define AS3 __attribute__((address_space(3)))
typedef int v4i_vs __attribute__((vector_size(4 * sizeof(int))));

// Builtin prototype (from clang diagnostic): (int4 AS1*, int4 AS3*, imm, imm).
// C++ disallows casts that change address space, so go through integers:
// generic->AS1 keeps the 64-bit address; generic->AS3 is addr[31:0] (LDS
// aperture truncation per the flat-addressing rules).
__device__ __forceinline__ void async_copy_b128(const void* gsrc, void* ldst) {
  __builtin_amdgcn_global_load_async_to_lds_b128(
      (AS1 v4i_vs*)(unsigned long long)(__SIZE_TYPE__)gsrc,
      (AS3 v4i_vs*)(unsigned int)(__SIZE_TYPE__)ldst,
      0, 0);
}
#endif

// ---------------------------------------------------------------------------
// Prep: fold BN scale into weights, emit f16 weights directly in WMMA A-frag
// layout: frag[(((mtile*9+tap)*2+kstep)*32+lane)*16+h], plus fused bias.
// A-frag K mapping (16-bit A 16x32): K = (lane<16?0:8) + (h&7) + (h>>3)*16.
// ---------------------------------------------------------------------------
__global__ __launch_bounds__(256) void prep_weights(
    const float* __restrict__ w, const float* __restrict__ bc,
    const float* __restrict__ g, const float* __restrict__ be,
    const float* __restrict__ m, const float* __restrict__ v,
    _Float16* __restrict__ frag, float* __restrict__ bias)
{
  int tid = blockIdx.x * 256 + threadIdx.x;        // 0..36863
  if (tid < 64) {
    float inv = g[tid] * rsqrtf(v[tid] + 1e-5f);
    bias[tid] = bc[tid] * inv + be[tid] - m[tid] * inv;
  }
  if (tid >= 4 * 9 * 2 * 32 * 16) return;
  int h     = tid & 15;
  int lane  = (tid >> 4) & 31;
  int kstep = (tid >> 9) & 1;
  int rest  = tid >> 10;                           // mtile*9 + tap
  int tap   = rest % 9;
  int mtile = rest / 9;
  int co    = mtile * 16 + (lane & 15);
  int kbase = (lane < 16) ? 0 : 8;
  int ci    = kstep * 32 + kbase + (h & 7) + ((h >> 3) << 4);
  int kh = tap / 3, kw = tap % 3;
  float inv = g[co] * rsqrtf(v[co] + 1e-5f);
  frag[tid] = (_Float16)(w[((co * 64 + ci) * 3 + kh) * 3 + kw] * inv);
}

// ---------------------------------------------------------------------------
// Half-resolution Haar edge: e = 0.5*(3a - b - c - d), shape (8,64,128,128)
// ---------------------------------------------------------------------------
__global__ __launch_bounds__(256) void edge_half_kernel(
    const float* __restrict__ x, float* __restrict__ e)
{
  int idx = blockIdx.x * 256 + threadIdx.x;        // 8*64*128*128
  int j  = idx & 127;
  int i  = (idx >> 7) & 127;
  int bc = idx >> 14;
  const float* r0 = x + (((size_t)bc * Hh + 2 * i) * Ww + 2 * j);
  float a = r0[0], b_ = r0[1], c_ = r0[Ww], d_ = r0[Ww + 1];
  e[idx] = 0.5f * (3.0f * a - b_ - c_ - d_);
}

// ---------------------------------------------------------------------------
// Conv1: x (f32 NCHW) -> implicit GEMM (9 shifted taps) -> bias -> PReLU
//        -> h (f16 NHWC)
// ---------------------------------------------------------------------------
__global__ __launch_bounds__(256) void conv1_kernel(
    const float* __restrict__ x, const _Float16* __restrict__ wf,
    const float* __restrict__ bias, const float* __restrict__ aprelu,
    _Float16* __restrict__ hout)
{
  __shared__ __align__(16) _Float16 lds[LROWS * LCOLS * PIXSTRIDE];
  const int t  = threadIdx.x;
  const int bb = blockIdx.z;
  const int r0 = blockIdx.y * TR;
  const int c0 = blockIdx.x * TC;

  // Pull the (L2-hot) weight fragments toward this WGP while we stage LDS.
  __builtin_prefetch(wf + t * 144, 0, 3);

  // Stage x tile (with halo) into LDS as [row][col][ci] f16, zero-padded.
  // Each thread owns a channel PAIR so LDS writes are packed b32 stores.
  for (int it = 0; it < 2; ++it) {
    int pair = t + it * 256;                       // (ci2, lrow): 320 pairs
    if (pair < 320) {
      int ci2  = pair & 31;
      int lrow = pair >> 5;
      int grow = r0 - 1 + lrow;
      if ((unsigned)grow < (unsigned)Hh) {
        const float* s0 = x + (((size_t)bb * Cc + ci2 * 2) * Hh + grow) * Ww;
        const float* s1 = s0 + (size_t)Hh * Ww;
        for (int lcol = 0; lcol < LCOLS; ++lcol) {
          int gcol = c0 - 1 + lcol;
          v2h pk{};
          if ((unsigned)gcol < (unsigned)Ww) {
            pk[0] = (_Float16)s0[gcol];
            pk[1] = (_Float16)s1[gcol];
          }
          *(v2h*)&lds[(lrow * LCOLS + lcol) * PIXSTRIDE + ci2 * 2] = pk;
        }
      } else {
        for (int lcol = 0; lcol < LCOLS; ++lcol) {
          v2h z{};
          *(v2h*)&lds[(lrow * LCOLS + lcol) * PIXSTRIDE + ci2 * 2] = z;
        }
      }
    }
  }
  __syncthreads();

  const int lane = t & 31;
  const int wv   = t >> 5;                         // tile row owned by wave
  const int nsel = lane & 15;
  const int bK   = (lane < 16) ? 0 : 8;            // B-frag K offset
  const _Float16* wbase = wf + (lane << 4);

  v8f acc[8] = {};

  for (int kh = 0; kh < 3; ++kh) {
    for (int kw = 0; kw < 3; ++kw) {
      const int tap = kh * 3 + kw;
      const int p0 = ((wv + kh) * LCOLS + (nsel + kw)) * PIXSTRIDE;
      const int p1 = p0 + 16 * PIXSTRIDE;
#pragma unroll
      for (int ks = 0; ks < 2; ++ks) {
        const int koff = ks * 32 + bK;
        V16 b0u, b1u;
        b0u.h[0] = *(const v8h*)&lds[p0 + koff];
        b0u.h[1] = *(const v8h*)&lds[p0 + koff + 16];
        b1u.h[0] = *(const v8h*)&lds[p1 + koff];
        b1u.h[1] = *(const v8h*)&lds[p1 + koff + 16];
#pragma unroll
        for (int m = 0; m < 4; ++m) {
          const v16h a = *(const v16h*)(wbase + (((m * 9 + tap) * 2 + ks) << 9));
          acc[m * 2 + 0] = __builtin_amdgcn_wmma_f32_16x16x32_f16(
              false, a, false, b0u.v, (short)0, acc[m * 2 + 0], false, false);
          acc[m * 2 + 1] = __builtin_amdgcn_wmma_f32_16x16x32_f16(
              false, a, false, b1u.v, (short)0, acc[m * 2 + 1], false, false);
        }
      }
    }
  }

  // Epilogue: fused bias + PReLU, store f16 NHWC.
  const float ap   = aprelu[0];
  const int hi8    = (lane >> 4) << 3;             // D: M = r + (lane>=16 ? 8:0)
  const int grow   = r0 + wv;
  const int gcol0  = c0 + nsel;
  _Float16* hrow = hout + (((size_t)bb * Hh + grow) * Ww) * 64;
#pragma unroll
  for (int m = 0; m < 4; ++m) {
#pragma unroll
    for (int r = 0; r < 8; ++r) {
      const int co = m * 16 + hi8 + r;
      const float bs = bias[co];
      float v0 = acc[m * 2 + 0][r] + bs;
      v0 = v0 > 0.0f ? v0 : ap * v0;
      hrow[(size_t)gcol0 * 64 + co] = (_Float16)v0;
      float v1 = acc[m * 2 + 1][r] + bs;
      v1 = v1 > 0.0f ? v1 : ap * v1;
      hrow[(size_t)(gcol0 + 16) * 64 + co] = (_Float16)v1;
    }
  }
}

// ---------------------------------------------------------------------------
// Conv2: h (f16 NHWC) -> implicit GEMM -> bias + x residual + bilinear(edge)
//        -> out (f32 NCHW). Staging uses async global->LDS copies (ASYNCcnt).
// ---------------------------------------------------------------------------
__global__ __launch_bounds__(256) void conv2_kernel(
    const _Float16* __restrict__ hin, const _Float16* __restrict__ wf,
    const float* __restrict__ bias, const float* __restrict__ x,
    const float* __restrict__ ebuf, float* __restrict__ out)
{
  __shared__ __align__(16) _Float16 lds[LROWS * LCOLS * PIXSTRIDE];
  const int t  = threadIdx.x;
  const int bb = blockIdx.z;
  const int r0 = blockIdx.y * TR;
  const int c0 = blockIdx.x * TC;

  __builtin_prefetch(wf + t * 144, 0, 3);

  // Stage h tile: 16B-chunk copies (NHWC -> LDS), zero halo.
  for (int it = 0; it < 11; ++it) {
    int job = t + it * 256;                        // 340 pixels * 8 chunks
    if (job < 2720) {
      int chunk = job & 7;
      int pix   = job >> 3;
      int lrow  = pix / LCOLS;
      int lcol  = pix - lrow * LCOLS;
      int grow = r0 - 1 + lrow, gcol = c0 - 1 + lcol;
      _Float16* dst = &lds[(lrow * LCOLS + lcol) * PIXSTRIDE + chunk * 8];
      if ((unsigned)grow < (unsigned)Hh && (unsigned)gcol < (unsigned)Ww) {
        const _Float16* src = hin +
            ((((size_t)bb * Hh + grow) * Ww + gcol) << 6) + chunk * 8;
#if defined(USE_ASYNC_LDS)
        async_copy_b128(src, dst);
#else
        *(v8h*)dst = *(const v8h*)src;
#endif
      } else {
        v8h z{};
        *(v8h*)dst = z;
      }
    }
  }
#if defined(USE_ASYNC_LDS)
  __builtin_amdgcn_s_wait_asynccnt(0);
#endif
  __syncthreads();

  const int lane = t & 31;
  const int wv   = t >> 5;
  const int nsel = lane & 15;
  const int bK   = (lane < 16) ? 0 : 8;
  const _Float16* wbase = wf + (lane << 4);

  v8f acc[8] = {};

  for (int kh = 0; kh < 3; ++kh) {
    for (int kw = 0; kw < 3; ++kw) {
      const int tap = kh * 3 + kw;
      const int p0 = ((wv + kh) * LCOLS + (nsel + kw)) * PIXSTRIDE;
      const int p1 = p0 + 16 * PIXSTRIDE;
#pragma unroll
      for (int ks = 0; ks < 2; ++ks) {
        const int koff = ks * 32 + bK;
        V16 b0u, b1u;
        b0u.h[0] = *(const v8h*)&lds[p0 + koff];
        b0u.h[1] = *(const v8h*)&lds[p0 + koff + 16];
        b1u.h[0] = *(const v8h*)&lds[p1 + koff];
        b1u.h[1] = *(const v8h*)&lds[p1 + koff + 16];
#pragma unroll
        for (int m = 0; m < 4; ++m) {
          const v16h a = *(const v16h*)(wbase + (((m * 9 + tap) * 2 + ks) << 9));
          acc[m * 2 + 0] = __builtin_amdgcn_wmma_f32_16x16x32_f16(
              false, a, false, b0u.v, (short)0, acc[m * 2 + 0], false, false);
          acc[m * 2 + 1] = __builtin_amdgcn_wmma_f32_16x16x32_f16(
              false, a, false, b1u.v, (short)0, acc[m * 2 + 1], false, false);
        }
      }
    }
  }

  // Epilogue: out = x + (conv2 + bias) + bilinear-upsampled half-res edge.
  const int hi8  = (lane >> 4) << 3;
  const int grow = r0 + wv;
  const int ca   = c0 + nsel;
  const int cb   = ca + 16;

  // Row bilinear setup (half-pixel centers, edge clamp), half-res = 128.
  float yc = grow * 0.5f - 0.25f;
  float y0f = floorf(yc);
  float fy = yc - y0f;
  int y0 = (int)y0f;
  int y0c = min(max(y0, 0), 127);
  int y1c = min(y0 + 1, 127);
  // Column setup for both N-tiles.
  float xca = ca * 0.5f - 0.25f, x0fa = floorf(xca), fxa = xca - x0fa;
  int xa0 = min(max((int)x0fa, 0), 127), xa1 = min((int)x0fa + 1, 127);
  float xcb = cb * 0.5f - 0.25f, x0fb = floorf(xcb), fxb = xcb - x0fb;
  int xb0 = min(max((int)x0fb, 0), 127), xb1 = min((int)x0fb + 1, 127);

#pragma unroll
  for (int m = 0; m < 4; ++m) {
#pragma unroll
    for (int r = 0; r < 8; ++r) {
      const int co = m * 16 + hi8 + r;
      const float bs = bias[co];
      const float* ec = ebuf + ((size_t)(bb * Cc + co) * 128) * 128;
      const float* e0 = ec + (size_t)y0c * 128;
      const float* e1 = ec + (size_t)y1c * 128;
      const float* xr = x + ((((size_t)bb * Cc + co) * Hh + grow) * Ww);
      float* outr = out + ((((size_t)bb * Cc + co) * Hh + grow) * Ww);

      float ea = (1.0f - fy) * ((1.0f - fxa) * e0[xa0] + fxa * e0[xa1]) +
                 fy         * ((1.0f - fxa) * e1[xa0] + fxa * e1[xa1]);
      float eb = (1.0f - fy) * ((1.0f - fxb) * e0[xb0] + fxb * e0[xb1]) +
                 fy         * ((1.0f - fxb) * e1[xb0] + fxb * e1[xb1]);

      outr[ca] = xr[ca] + (acc[m * 2 + 0][r] + bs) + ea;
      outr[cb] = xr[cb] + (acc[m * 2 + 1][r] + bs) + eb;
    }
  }
}

// ---------------------------------------------------------------------------
extern "C" void kernel_launch(void* const* d_in, const int* in_sizes, int n_in,
                              void* d_out, int out_size, void* d_ws, size_t ws_size,
                              hipStream_t stream) {
  const float* x   = (const float*)d_in[0];
  const float* w1  = (const float*)d_in[1];
  const float* b1  = (const float*)d_in[2];
  const float* g1  = (const float*)d_in[3];
  const float* be1 = (const float*)d_in[4];
  const float* m1  = (const float*)d_in[5];
  const float* v1  = (const float*)d_in[6];
  const float* ap  = (const float*)d_in[7];
  const float* w2  = (const float*)d_in[8];
  const float* b2  = (const float*)d_in[9];
  const float* g2  = (const float*)d_in[10];
  const float* be2 = (const float*)d_in[11];
  const float* m2  = (const float*)d_in[12];
  const float* v2  = (const float*)d_in[13];

  char* ws = (char*)d_ws;
  _Float16* wf1   = (_Float16*)(ws + 0);              // 73728 B
  _Float16* wf2   = (_Float16*)(ws + 73728);          // 73728 B
  float*    bias1 = (float*)(ws + 147456);            // 256 B
  float*    bias2 = (float*)(ws + 147712);            // 256 B
  _Float16* hbuf  = (_Float16*)(ws + (1ull << 22));   // 64 MB (NHWC f16)
  float*    ebuf  = (float*)(ws + 71303168ull);       // 32 MB (half-res edge)

  prep_weights<<<144, 256, 0, stream>>>(w1, b1, g1, be1, m1, v1, wf1, bias1);
  prep_weights<<<144, 256, 0, stream>>>(w2, b2, g2, be2, m2, v2, wf2, bias2);
  edge_half_kernel<<<(Bn * Cc * 128 * 128) / 256, 256, 0, stream>>>(x, ebuf);

  dim3 grid(Ww / TC, Hh / TR, Bn);
  conv1_kernel<<<grid, 256, 0, stream>>>(x, wf1, bias1, ap, hbuf);
  conv2_kernel<<<grid, 256, 0, stream>>>(hbuf, wf2, bias2, x, ebuf, (float*)d_out);
}
